// MediumASTGCN_39608188404351
// MI455X (gfx1250) — matrix operand order repaired and checked by reading, not compiled, not run on hardware
//
#include <hip/hip_runtime.h>
#include <math.h>

// ---------------------------------------------------------------------------
// MediumASTGCN forward for MI455X (gfx1250, wave32).
// Heavy all-pairs edge MLP (13 GFLOP of the ~17 GFLOP total) runs as a fused
// bf16 WMMA kernel (v_wmma_f32_16x16x32_bf16) with in-register LayerNorm/GELU.
// All small einsums / convs / attentions are VALU kernels (latency bound).
// ---------------------------------------------------------------------------

enum { Bn = 4, Tn = 10, Nn = 256, Cn = 6, Hn = 64, En = 4096 };
#define LN_EPS 1e-5f
#define BN_EPS 1e-5f
#define XSZ (Bn * Nn * Hn * Tn) /* 655360 */

typedef __attribute__((ext_vector_type(16))) __bf16 v16bf;
typedef __attribute__((ext_vector_type(8))) float v8f;

union FragBF {
  v16bf v;
  uint4 u[2];
};

#define DEV static __device__ __forceinline__

DEV float geluf(float x) { return 0.5f * x * (1.f + erff(x * 0.70710678118654752440f)); }
DEV float sigm(float x) { return 1.f / (1.f + expf(-x)); }
DEV unsigned short f2bf(float f) {
  __bf16 h = (__bf16)f;
  unsigned short u;
  __builtin_memcpy(&u, &h, 2);
  return u;
}

// ---------------------------------------------------------------------------
// utility kernels
// ---------------------------------------------------------------------------
__global__ void k_zero_f(float* p, int n) {
  int i = blockIdx.x * 256 + threadIdx.x;
  if (i < n) p[i] = 0.f;
}
__global__ void k_zero_i(int* p, int n) {
  int i = blockIdx.x * 256 + threadIdx.x;
  if (i < n) p[i] = 0;
}
__global__ void k_deg(const int* __restrict__ ei, int* __restrict__ deg) {
  int e = blockIdx.x * 256 + threadIdx.x;
  if (e < En) atomicAdd(&deg[ei[e]], 1);
}
__global__ void k_deginfo(const int* __restrict__ deg, const float* __restrict__ emb,
                          float* __restrict__ deginv, float* __restrict__ dmean) {
  int n = threadIdx.x;
  if (n >= Nn) return;
  int d = deg[n];
  deginv[n] = (d > 0) ? 1.f / (float)d : 0.f;
  int dc = d < (Nn - 1) ? d : (Nn - 1);
  float s = 0.f;
  for (int h = 0; h < Hn; ++h) s += emb[dc * Hn + h];
  dmean[n] = s * (1.f / (float)Hn);
}
__global__ void k_cvt_bf(const float* __restrict__ in, unsigned short* __restrict__ out, int n) {
  int i = blockIdx.x * 256 + threadIdx.x;
  if (i < n) out[i] = f2bf(in[i]);
}

// ---------------------------------------------------------------------------
// dilated temporal conv stack (B,C,T,N)->(B,H,T,N), BN(eval) + exact GELU
// ---------------------------------------------------------------------------
__global__ void k_conv1(const float* __restrict__ x, const float* __restrict__ w,
                        const float* __restrict__ bias, const float* __restrict__ bng,
                        const float* __restrict__ bnb, float* __restrict__ out) {
  int idx = blockIdx.x * 256 + threadIdx.x;
  if (idx >= Bn * Hn * Tn * Nn) return;
  int n = idx % Nn, t = (idx / Nn) % Tn, o = (idx / (Nn * Tn)) % Hn, b = idx / (Nn * Tn * Hn);
  float acc = bias[o];
  for (int dt = 0; dt < 3; ++dt) {
    int ti = t - 1 + dt;
    if (ti < 0 || ti >= Tn) continue;
    for (int c = 0; c < Cn; ++c)
      acc += x[((b * Tn + ti) * Nn + n) * Cn + c] * w[(o * Cn + c) * 3 + dt];
  }
  float v = acc * (bng[o] * rsqrtf(1.f + BN_EPS)) + bnb[o];
  out[idx] = geluf(v);
}
__global__ void k_convH(const float* __restrict__ in, const float* __restrict__ w,
                        const float* __restrict__ bias, const float* __restrict__ bng,
                        const float* __restrict__ bnb, float* __restrict__ out, int dil, int pad) {
  int idx = blockIdx.x * 256 + threadIdx.x;
  if (idx >= Bn * Hn * Tn * Nn) return;
  int n = idx % Nn, t = (idx / Nn) % Tn, o = (idx / (Nn * Tn)) % Hn, b = idx / (Nn * Tn * Hn);
  float acc = bias[o];
  for (int dt = 0; dt < 3; ++dt) {
    int ti = t - pad + dil * dt;
    if (ti < 0 || ti >= Tn) continue;
    const float* ip = in + ((size_t)(b * Hn) * Tn + ti) * Nn + n;
    const float* wp = w + (o * Hn) * 3 + dt;
    for (int c = 0; c < Hn; ++c) acc += ip[(size_t)c * Tn * Nn] * wp[c * 3];
  }
  float v = acc * (bng[o] * rsqrtf(1.f + BN_EPS)) + bnb[o];
  out[idx] = geluf(v);
}
// (B,H,T,N) -> X (B,N,H,T)
__global__ void k_transX(const float* __restrict__ cv, float* __restrict__ X) {
  int idx = blockIdx.x * 256 + threadIdx.x;
  if (idx >= XSZ) return;
  int t = idx % Tn, f = (idx / Tn) % Hn, n = (idx / (Tn * Hn)) % Nn, b = idx / (Tn * Hn * Nn);
  X[idx] = cv[((b * Hn + f) * Tn + t) * Nn + n];
}

// ---------------------------------------------------------------------------
// ASTGCN block kernels.  X layout: (B,N,F=64,T=10)
// ---------------------------------------------------------------------------
__global__ void k_tmp1(const float* __restrict__ X, const float* __restrict__ U1,
                       float* __restrict__ tmp1) {
  int idx = blockIdx.x * 256 + threadIdx.x;
  if (idx >= Bn * Hn * Tn) return;
  int t = idx % Tn, f = (idx / Tn) % Hn, b = idx / (Tn * Hn);
  float s = 0.f;
  for (int n = 0; n < Nn; ++n) s += X[((size_t)(b * Nn + n) * Hn + f) * Tn + t] * U1[n];
  tmp1[(b * Hn + f) * Tn + t] = s;
}
__global__ void k_lhsT(const float* __restrict__ tmp1, const float* __restrict__ U2,
                       float* __restrict__ lhs) {
  int idx = blockIdx.x * 256 + threadIdx.x;
  if (idx >= Bn * Tn * Nn) return;
  int m = idx % Nn, t = (idx / Nn) % Tn, b = idx / (Nn * Tn);
  float s = 0.f;
  for (int f = 0; f < Hn; ++f) s += tmp1[(b * Hn + f) * Tn + t] * U2[f * Nn + m];
  lhs[(b * Tn + t) * Nn + m] = s;
}
__global__ void k_rhsT(const float* __restrict__ X, const float* __restrict__ U3,
                       float* __restrict__ rhs) {
  int idx = blockIdx.x * 256 + threadIdx.x;
  if (idx >= Bn * Nn * Tn) return;
  int t = idx % Tn, n = (idx / Tn) % Nn, b = idx / (Tn * Nn);
  float s = 0.f;
  for (int f = 0; f < Hn; ++f) s += U3[f] * X[((size_t)(b * Nn + n) * Hn + f) * Tn + t];
  rhs[(b * Nn + n) * Tn + t] = s;
}
__global__ void k_p1(const float* __restrict__ lhs, const float* __restrict__ rhs,
                     const float* __restrict__ be, float* __restrict__ p1) {
  int idx = blockIdx.x * 256 + threadIdx.x;
  if (idx >= Bn * Tn * Tn) return;
  int u = idx % Tn, t = (idx / Tn) % Tn, b = idx / (Tn * Tn);
  float s = 0.f;
  for (int n = 0; n < Nn; ++n) s += lhs[(b * Tn + t) * Nn + n] * rhs[(b * Nn + n) * Tn + u];
  p1[(b * Tn + t) * Tn + u] = sigm(s + be[t * Tn + u]);
}
__global__ void k_eatt(const float* __restrict__ Ve, const float* __restrict__ p1,
                       float* __restrict__ e) {
  int idx = blockIdx.x * 256 + threadIdx.x;
  if (idx >= Bn * Tn * Tn) return;
  int t = idx % Tn, u = (idx / Tn) % Tn, b = idx / (Tn * Tn);
  float s = 0.f;
  for (int v = 0; v < Tn; ++v) s += Ve[u * Tn + v] * p1[(b * Tn + v) * Tn + t];
  e[(b * Tn + u) * Tn + t] = s;
}
__global__ void k_smT(float* __restrict__ e) {  // softmax over axis 1 of (B,T,T)
  int idx = blockIdx.x * 256 + threadIdx.x;
  if (idx >= Bn * Tn) return;
  int t = idx % Tn, b = idx / Tn;
  float mx = -1e30f;
  for (int u = 0; u < Tn; ++u) mx = fmaxf(mx, e[(b * Tn + u) * Tn + t]);
  float s = 0.f;
  for (int u = 0; u < Tn; ++u) s += expf(e[(b * Tn + u) * Tn + t] - mx);
  float inv = 1.f / s;
  for (int u = 0; u < Tn; ++u) {
    float* p = &e[(b * Tn + u) * Tn + t];
    *p = expf(*p - mx) * inv;
  }
}
__global__ void k_Xt(const float* __restrict__ X, const float* __restrict__ e,
                     float* __restrict__ Xt) {
  int idx = blockIdx.x * 256 + threadIdx.x;
  if (idx >= XSZ) return;
  int t = idx % Tn, f = (idx / Tn) % Hn, n = (idx / (Tn * Hn)) % Nn, b = idx / (Tn * Hn * Nn);
  float s = 0.f;
  for (int u = 0; u < Tn; ++u)
    s += X[((size_t)(b * Nn + n) * Hn + f) * Tn + u] * e[(b * Tn + u) * Tn + t];
  Xt[idx] = s;
}
__global__ void k_tmp2(const float* __restrict__ Xt, const float* __restrict__ W1,
                       float* __restrict__ tmp2) {
  int idx = blockIdx.x * 256 + threadIdx.x;
  if (idx >= Bn * Nn * Hn) return;
  int f = idx % Hn, n = (idx / Hn) % Nn, b = idx / (Hn * Nn);
  float s = 0.f;
  for (int t = 0; t < Tn; ++t) s += Xt[((size_t)(b * Nn + n) * Hn + f) * Tn + t] * W1[t];
  tmp2[(b * Nn + n) * Hn + f] = s;
}
__global__ void k_lhsS(const float* __restrict__ tmp2, const float* __restrict__ W2,
                       float* __restrict__ lhs) {
  int idx = blockIdx.x * 256 + threadIdx.x;
  if (idx >= Bn * Nn * Tn) return;
  int m = idx % Tn, n = (idx / Tn) % Nn, b = idx / (Tn * Nn);
  float s = 0.f;
  for (int f = 0; f < Hn; ++f) s += tmp2[(b * Nn + n) * Hn + f] * W2[f * Tn + m];
  lhs[(b * Nn + n) * Tn + m] = s;
}
__global__ void k_rhsS(const float* __restrict__ Xt, const float* __restrict__ W3,
                       float* __restrict__ rhs) {
  int idx = blockIdx.x * 256 + threadIdx.x;
  if (idx >= Bn * Tn * Nn) return;
  int n = idx % Nn, t = (idx / Nn) % Tn, b = idx / (Nn * Tn);
  float s = 0.f;
  for (int f = 0; f < Hn; ++f) s += W3[f] * Xt[((size_t)(b * Nn + n) * Hn + f) * Tn + t];
  rhs[(b * Tn + t) * Nn + n] = s;
}
__global__ void k_sPre(const float* __restrict__ lhs, const float* __restrict__ rhs,
                       const float* __restrict__ bs, float* __restrict__ sp) {
  int idx = blockIdx.x * 256 + threadIdx.x;
  if (idx >= Bn * Nn * Nn) return;
  int m = idx % Nn, n = (idx / Nn) % Nn, b = idx / (Nn * Nn);
  float s = 0.f;
  for (int t = 0; t < Tn; ++t) s += lhs[(b * Nn + n) * Tn + t] * rhs[(b * Tn + t) * Nn + m];
  sp[((size_t)(b * Nn + n)) * Nn + m] = sigm(s + bs[n * Nn + m]);
}
__global__ void k_Svs(const float* __restrict__ Vs, const float* __restrict__ sp,
                      float* __restrict__ S) {
  int idx = blockIdx.x * 256 + threadIdx.x;
  if (idx >= Bn * Nn * Nn) return;
  int n = idx % Nn, u = (idx / Nn) % Nn, b = idx / (Nn * Nn);
  float s = 0.f;
  for (int v = 0; v < Nn; ++v) s += Vs[u * Nn + v] * sp[((size_t)(b * Nn + v)) * Nn + n];
  S[((size_t)(b * Nn + u)) * Nn + n] = s;
}
__global__ void k_smS(float* __restrict__ S) {  // softmax over axis 1 of (B,N,N)
  int idx = blockIdx.x * 256 + threadIdx.x;
  if (idx >= Bn * Nn) return;
  int n = idx % Nn, b = idx / Nn;
  float mx = -1e30f;
  for (int u = 0; u < Nn; ++u) mx = fmaxf(mx, S[((size_t)(b * Nn + u)) * Nn + n]);
  float s = 0.f;
  for (int u = 0; u < Nn; ++u) s += expf(S[((size_t)(b * Nn + u)) * Nn + n] - mx);
  float inv = 1.f / s;
  for (int u = 0; u < Nn; ++u) {
    float* p = &S[((size_t)(b * Nn + u)) * Nn + n];
    *p = expf(*p - mx) * inv;
  }
}
__global__ void k_tax0(const float* __restrict__ X, const float* __restrict__ S,
                       float* __restrict__ tax0) {
  int idx = blockIdx.x * 256 + threadIdx.x;
  if (idx >= XSZ) return;
  int n = (idx / (Tn * Hn)) % Nn, b = idx / (Tn * Hn * Nn);
  tax0[idx] = X[idx] * S[((size_t)(b * Nn + n)) * Nn + n];
}
__global__ void k_scatter(const int* __restrict__ ei, const float* __restrict__ deginv,
                          const float* __restrict__ S, const float* __restrict__ tax0,
                          float* __restrict__ tax1) {
  int idx = blockIdx.x * 256 + threadIdx.x;
  const int FT = Hn * Tn;  // 640
  if (idx >= Bn * En * FT) return;
  int k = idx % FT, e = (idx / FT) % En, b = idx / (FT * En);
  int s = ei[e], d = ei[En + e];
  float coef = -deginv[s] * S[((size_t)(b * Nn + s)) * Nn + d];
  atomicAdd(&tax1[(size_t)(b * Nn + s) * FT + k], coef * tax0[(size_t)(b * Nn + d) * FT + k]);
}
__global__ void k_sg(const float* __restrict__ tax0, const float* __restrict__ tax1,
                     const float* __restrict__ chebw, const float* __restrict__ chebb,
                     float* __restrict__ sg) {
  int idx = blockIdx.x * 256 + threadIdx.x;
  if (idx >= XSZ) return;
  int t = idx % Tn, g = (idx / Tn) % Hn, n = (idx / (Tn * Hn)) % Nn, b = idx / (Tn * Hn * Nn);
  const float* W0 = chebw;
  const float* W1c = chebw + Hn * Hn;
  size_t base = (size_t)(b * Nn + n) * (Hn * Tn);
  float s = chebb[g];
  for (int f = 0; f < Hn; ++f) {
    int ft = f * Tn + t;
    s += tax0[base + ft] * W0[f * Hn + g] + tax1[base + ft] * W1c[f * Hn + g];
  }
  sg[idx] = fmaxf(s, 0.f);
}
__global__ void k_tconv(const float* __restrict__ sg, const float* __restrict__ tw,
                        const float* __restrict__ tb, float* __restrict__ out) {
  int idx = blockIdx.x * 256 + threadIdx.x;
  if (idx >= Bn * Hn * Nn * Tn) return;
  int t = idx % Tn, n = (idx / Tn) % Nn, o = (idx / (Tn * Nn)) % Hn, b = idx / (Tn * Nn * Hn);
  float s = tb[o];
  for (int dt = 0; dt < 3; ++dt) {
    int tt = t - 1 + dt;
    if (tt < 0 || tt >= Tn) continue;
    for (int g = 0; g < Hn; ++g)
      s += sg[((size_t)(b * Nn + n) * Hn + g) * Tn + tt] * tw[(o * Hn + g) * 3 + dt];
  }
  out[idx] = s;  // (B,H,N,T)
}
__global__ void k_res(const float* __restrict__ X, const float* __restrict__ rw,
                      const float* __restrict__ rb, float* __restrict__ out) {
  int idx = blockIdx.x * 256 + threadIdx.x;
  if (idx >= Bn * Hn * Nn * Tn) return;
  int t = idx % Tn, n = (idx / Tn) % Nn, o = (idx / (Tn * Nn)) % Hn, b = idx / (Tn * Nn * Hn);
  float s = rb[o];
  for (int f = 0; f < Hn; ++f) s += X[((size_t)(b * Nn + n) * Hn + f) * Tn + t] * rw[o * Hn + f];
  out[idx] = s;  // (B,H,N,T)
}
__global__ void k_lnout(const float* __restrict__ res, const float* __restrict__ tc,
                        const float* __restrict__ lng, const float* __restrict__ lnb,
                        float* __restrict__ Xout) {
  int idx = blockIdx.x * 256 + threadIdx.x;
  if (idx >= Bn * Tn * Nn) return;
  int n = idx % Nn, t = (idx / Nn) % Tn, b = idx / (Nn * Tn);
  float s = 0.f, ss = 0.f;
  for (int g = 0; g < Hn; ++g) {
    size_t p = ((size_t)(b * Hn + g) * Nn + n) * Tn + t;
    float v = fmaxf(res[p] + tc[p], 0.f);
    s += v;
    ss += v * v;
  }
  float mean = s * (1.f / (float)Hn);
  float inv = rsqrtf(ss * (1.f / (float)Hn) - mean * mean + LN_EPS);
  for (int g = 0; g < Hn; ++g) {
    size_t p = ((size_t)(b * Hn + g) * Nn + n) * Tn + t;
    float v = fmaxf(res[p] + tc[p], 0.f);
    Xout[((size_t)(b * Nn + n) * Hn + g) * Tn + t] = (v - mean) * inv * lng[g] + lnb[g];
  }
}

// ---------------------------------------------------------------------------
// tail: final conv, node MLP, nf build
// ---------------------------------------------------------------------------
__global__ void k_gf(const float* __restrict__ X, const float* __restrict__ fcw,
                     const float* __restrict__ fcb, float* __restrict__ gf) {
  int idx = blockIdx.x * 256 + threadIdx.x;
  if (idx >= Bn * Nn) return;
  int n = idx % Nn, b = idx / Nn;
  float s = fcb[0];
  for (int t = 0; t < Tn; ++t)
    for (int g = 0; g < Hn; ++g)
      s += X[((size_t)(b * Nn + n) * Hn + g) * Tn + t] * fcw[t * Hn + g];
  gf[idx] = s;
}
__global__ void k_nlin1(const float* __restrict__ gf, const float* __restrict__ w1,
                        const float* __restrict__ b1, float* __restrict__ out) {
  int idx = blockIdx.x * 256 + threadIdx.x;
  if (idx >= Bn * Nn * Hn) return;
  int h = idx % Hn, r = idx / Hn;
  out[idx] = gf[r] * w1[h] + b1[h];
}
__global__ void k_lin(const float* __restrict__ in, const float* __restrict__ w,
                      const float* __restrict__ bias, float* __restrict__ out, int din, int dout,
                      int rows) {
  int idx = blockIdx.x * 256 + threadIdx.x;
  if (idx >= rows * dout) return;
  int o = idx % dout, r = idx / dout;
  const float* ip = in + (size_t)r * din;
  const float* wp = w + (size_t)o * din;
  float s = bias[o];
  for (int k = 0; k < din; ++k) s += ip[k] * wp[k];
  out[idx] = s;
}
__global__ void k_ln_rows(float* __restrict__ buf, const float* __restrict__ g,
                          const float* __restrict__ b, int rows, int dim, int dogelu) {
  int r = blockIdx.x * 256 + threadIdx.x;
  if (r >= rows) return;
  float* p = buf + (size_t)r * dim;
  float s = 0.f, ss = 0.f;
  for (int d = 0; d < dim; ++d) {
    float v = p[d];
    s += v;
    ss += v * v;
  }
  float mean = s / (float)dim;
  float inv = rsqrtf(ss / (float)dim - mean * mean + LN_EPS);
  for (int d = 0; d < dim; ++d) {
    float v = (p[d] - mean) * inv * g[d] + b[d];
    p[d] = dogelu ? geluf(v) : v;
  }
}
__global__ void k_nf_build(const float* __restrict__ nm, const float* __restrict__ emb,
                           const int* __restrict__ deg, unsigned short* __restrict__ nfb) {
  int idx = blockIdx.x * 256 + threadIdx.x;
  if (idx >= Bn * Nn * Hn) return;
  int h = idx % Hn, n = (idx / Hn) % Nn;
  int dc = deg[n] < (Nn - 1) ? deg[n] : (Nn - 1);
  nfb[idx] = f2bf(nm[idx] + emb[dc * Hn + h]);
}

// ---------------------------------------------------------------------------
// Fused all-pairs edge MLP: 128 -> LN,GELU -> 128x128 -> LN,GELU(64) -> 1
// One wave per 16-pair tile (fixed i, 16 consecutive j).  48 WMMAs / wave.
// ---------------------------------------------------------------------------
__global__ __launch_bounds__(32) void k_edge(
    const unsigned short* __restrict__ nfb_u, const unsigned short* __restrict__ w1b_u,
    const float* __restrict__ b1, const float* __restrict__ g1, const float* __restrict__ be1,
    const unsigned short* __restrict__ w2b_u, const float* __restrict__ b2,
    const float* __restrict__ g2, const float* __restrict__ be2, const float* __restrict__ w3,
    const float* __restrict__ b3, const float* __restrict__ dmean, float* __restrict__ Araw) {
  const __bf16* nfb = (const __bf16*)nfb_u;
  const __bf16* w1b = (const __bf16*)w1b_u;
  const __bf16* w2b = (const __bf16*)w2b_u;
  const int lane = threadIdx.x;
  const int lane16 = lane & 15;
  const int hi = lane >> 4;
  const int j0 = blockIdx.x * 16;
  const int i = blockIdx.y;
  const int b = blockIdx.z;
  __shared__ __bf16 h1s[16 * 128];

  // ---- layer 1 A fragments (pair inputs): k<64 -> nf[i], k>=64 -> nf[j0+m]
  FragBF afr[4];
  const __bf16* rowi = nfb + ((size_t)(b * Nn + i)) * Hn;
  const __bf16* rowj = nfb + ((size_t)(b * Nn + j0 + lane16)) * Hn;
#pragma unroll
  for (int kc = 0; kc < 4; ++kc) {
    const __bf16* src = (kc < 2) ? rowi : rowj;
    int base = (kc & 1) * 32 + hi * 8;
    afr[kc].u[0] = *(const uint4*)(src + base);
    afr[kc].u[1] = *(const uint4*)(src + base + 16);
  }

  // ---- layer 1: 8 N-tiles x 4 K-chunks of v_wmma_f32_16x16x32_bf16
  v8f acc[8];
#pragma unroll
  for (int nt = 0; nt < 8; ++nt) {
    float bv = b1[nt * 16 + lane16];
#pragma unroll
    for (int r = 0; r < 8; ++r) acc[nt][r] = bv;
  }
#pragma unroll
  for (int nt = 0; nt < 8; ++nt) {
    const __bf16* wrow = w1b + (size_t)(nt * 16 + lane16) * 128 + hi * 8;
#pragma unroll
    for (int kc = 0; kc < 4; ++kc) {
      FragBF bf;
      const __bf16* p = wrow + kc * 32;
      bf.u[0] = *(const uint4*)p;
      bf.u[1] = *(const uint4*)(p + 16);
      acc[nt] = __builtin_amdgcn_wmma_f32_16x16x32_bf16(false, afr[kc].v, false, bf.v, (short)0,
                                                        acc[nt], false, false);
    }
  }

  // ---- LN(128) + GELU in register, re-swizzle via LDS into A-fragment layout
#pragma unroll
  for (int r = 0; r < 8; ++r) {
    float s = 0.f, ss = 0.f;
#pragma unroll
    for (int nt = 0; nt < 8; ++nt) {
      float v = acc[nt][r];
      s += v;
      ss += v * v;
    }
    s += __shfl_xor(s, 1, 32);  ss += __shfl_xor(ss, 1, 32);
    s += __shfl_xor(s, 2, 32);  ss += __shfl_xor(ss, 2, 32);
    s += __shfl_xor(s, 4, 32);  ss += __shfl_xor(ss, 4, 32);
    s += __shfl_xor(s, 8, 32);  ss += __shfl_xor(ss, 8, 32);
    float mean = s * (1.f / 128.f);
    float inv = rsqrtf(ss * (1.f / 128.f) - mean * mean + LN_EPS);
    int mrow = r + hi * 8;
#pragma unroll
    for (int nt = 0; nt < 8; ++nt) {
      int nn = nt * 16 + lane16;
      float v = (acc[nt][r] - mean) * inv * g1[nn] + be1[nn];
      h1s[mrow * 128 + nn] = (__bf16)geluf(v);
    }
  }
  __syncthreads();

  // ---- layer 2: 4 N-tiles x 4 K-chunks
  FragBF a2[4];
#pragma unroll
  for (int kc = 0; kc < 4; ++kc) {
    const __bf16* p = &h1s[lane16 * 128 + kc * 32 + hi * 8];
    a2[kc].u[0] = *(const uint4*)p;
    a2[kc].u[1] = *(const uint4*)(p + 16);
  }
  v8f acc2[4];
#pragma unroll
  for (int nt = 0; nt < 4; ++nt) {
    float bv = b2[nt * 16 + lane16];
#pragma unroll
    for (int r = 0; r < 8; ++r) acc2[nt][r] = bv;
  }
#pragma unroll
  for (int nt = 0; nt < 4; ++nt) {
    const __bf16* wrow = w2b + (size_t)(nt * 16 + lane16) * 128 + hi * 8;
#pragma unroll
    for (int kc = 0; kc < 4; ++kc) {
      FragBF bf;
      const __bf16* p = wrow + kc * 32;
      bf.u[0] = *(const uint4*)p;
      bf.u[1] = *(const uint4*)(p + 16);
      acc2[nt] = __builtin_amdgcn_wmma_f32_16x16x32_bf16(false, a2[kc].v, false, bf.v, (short)0,
                                                         acc2[nt], false, false);
    }
  }

  // ---- LN(64) + GELU + layer-3 dot(64) + bias/output
  float b3v = b3[0];
  float dmi = dmean[i];
#pragma unroll
  for (int r = 0; r < 8; ++r) {
    float s = 0.f, ss = 0.f;
#pragma unroll
    for (int nt = 0; nt < 4; ++nt) {
      float v = acc2[nt][r];
      s += v;
      ss += v * v;
    }
    s += __shfl_xor(s, 1, 32);  ss += __shfl_xor(ss, 1, 32);
    s += __shfl_xor(s, 2, 32);  ss += __shfl_xor(ss, 2, 32);
    s += __shfl_xor(s, 4, 32);  ss += __shfl_xor(ss, 4, 32);
    s += __shfl_xor(s, 8, 32);  ss += __shfl_xor(ss, 8, 32);
    float mean = s * (1.f / 64.f);
    float inv = rsqrtf(ss * (1.f / 64.f) - mean * mean + LN_EPS);
    float part = 0.f;
#pragma unroll
    for (int nt = 0; nt < 4; ++nt) {
      int nn = nt * 16 + lane16;
      float v = geluf((acc2[nt][r] - mean) * inv * g2[nn] + be2[nn]);
      part += v * w3[nn];
    }
    part += __shfl_xor(part, 1, 32);
    part += __shfl_xor(part, 2, 32);
    part += __shfl_xor(part, 4, 32);
    part += __shfl_xor(part, 8, 32);
    if (lane16 == 0) {
      int mrow = r + hi * 8;
      int j = j0 + mrow;
      float ep = (part + b3v) * 1.5f;
      float df = sigm(dmi * dmean[j]);
      float bias = (df > 0.5f) ? 0.f : -(1.f - df);
      Araw[((size_t)(b * Nn + i)) * Nn + j] = ep + bias;
    }
  }
}

__global__ void k_final(const float* __restrict__ Araw, float* __restrict__ out) {
  int idx = blockIdx.x * 256 + threadIdx.x;
  if (idx >= Bn * Nn * Nn) return;
  int j = idx % Nn, i = (idx / Nn) % Nn, b = idx / (Nn * Nn);
  if (i == j) {
    out[idx] = -2.f;
  } else {
    out[idx] =
        0.5f * (Araw[((size_t)(b * Nn + i)) * Nn + j] + Araw[((size_t)(b * Nn + j)) * Nn + i]);
  }
}

// ---------------------------------------------------------------------------
// host driver
// ---------------------------------------------------------------------------
#define GRID(n) dim3((unsigned)(((n) + 255) / 256)), dim3(256), 0, stream

extern "C" void kernel_launch(void* const* d_in, const int* in_sizes, int n_in, void* d_out,
                              int out_size, void* d_ws, size_t ws_size, hipStream_t stream) {
  (void)in_sizes;
  (void)n_in;
  (void)out_size;
  (void)ws_size;
  const float* xin = (const float*)d_in[0];
  const int* ei = (const int*)d_in[1];
#define F(i) ((const float*)d_in[(i)])

  float* ws = (float*)d_ws;
  size_t off = 0;
  auto alloc = [&](size_t nfl) {
    float* p = ws + off;
    off += nfl;
    return p;
  };
  float* XA = alloc(XSZ);
  float* XB = alloc(XSZ);
  float* CV1 = alloc(XSZ);
  float* CV2 = alloc(XSZ);
  float* tmp1 = alloc(Bn * Hn * Tn);
  float* lhsT = alloc(Bn * Tn * Nn);
  float* rhsT = alloc(Bn * Nn * Tn);
  float* p1 = alloc(Bn * Tn * Tn);
  float* eA = alloc(Bn * Tn * Tn);
  float* Xt = alloc(XSZ);
  float* tmp2 = alloc(Bn * Nn * Hn);
  float* lhsS = alloc(Bn * Nn * Tn);
  float* rhsS = alloc(Bn * Tn * Nn);
  float* sPre = alloc((size_t)Bn * Nn * Nn);
  float* Smat = alloc((size_t)Bn * Nn * Nn);
  float* tax0 = alloc(XSZ);
  float* tax1 = alloc(XSZ);
  float* sg = alloc(XSZ);
  float* tcb = alloc(XSZ);
  float* resb = alloc(XSZ);
  float* gf = alloc(Bn * Nn);
  float* nmA = alloc(Bn * Nn * Hn);
  float* nmB = alloc(Bn * Nn * Hn);
  int* deg = (int*)alloc(Nn);
  float* deginv = alloc(Nn);
  float* dmean = alloc(Nn);
  float* Araw = alloc((size_t)Bn * Nn * Nn);
  unsigned short* nfb = (unsigned short*)alloc(Bn * Nn * Hn / 2);
  unsigned short* w1b = (unsigned short*)alloc(128 * 128 / 2);
  unsigned short* w2b = (unsigned short*)alloc(64 * 128 / 2);

  // degree info
  k_zero_i<<<GRID(Nn)>>>(deg, Nn);
  k_deg<<<GRID(En)>>>(ei, deg);
  k_deginfo<<<dim3(1), dim3(256), 0, stream>>>(deg, F(74), deginv, dmean);

  // dilated temporal conv stack
  k_conv1<<<GRID(XSZ)>>>(xin, F(2), F(3), F(4), F(5), CV1);
  k_convH<<<GRID(XSZ)>>>(CV1, F(6), F(7), F(8), F(9), CV2, 2, 2);
  k_convH<<<GRID(XSZ)>>>(CV2, F(10), F(11), F(12), F(13), CV1, 4, 4);
  k_transX<<<GRID(XSZ)>>>(CV1, XA);

  // two ASTGCN blocks (params at base 14 and 32), ping-pong XA/XB
  auto run_block = [&](int base, const float* Xin, float* Xout) {
    k_tmp1<<<GRID(Bn * Hn * Tn)>>>(Xin, F(base + 0), tmp1);
    k_lhsT<<<GRID(Bn * Tn * Nn)>>>(tmp1, F(base + 1), lhsT);
    k_rhsT<<<GRID(Bn * Nn * Tn)>>>(Xin, F(base + 2), rhsT);
    k_p1<<<GRID(Bn * Tn * Tn)>>>(lhsT, rhsT, F(base + 3), p1);
    k_eatt<<<GRID(Bn * Tn * Tn)>>>(F(base + 4), p1, eA);
    k_smT<<<GRID(Bn * Tn)>>>(eA);
    k_Xt<<<GRID(XSZ)>>>(Xin, eA, Xt);
    k_tmp2<<<GRID(Bn * Nn * Hn)>>>(Xt, F(base + 5), tmp2);
    k_lhsS<<<GRID(Bn * Nn * Tn)>>>(tmp2, F(base + 6), lhsS);
    k_rhsS<<<GRID(Bn * Tn * Nn)>>>(Xt, F(base + 7), rhsS);
    k_sPre<<<GRID(Bn * Nn * Nn)>>>(lhsS, rhsS, F(base + 8), sPre);
    k_Svs<<<GRID(Bn * Nn * Nn)>>>(F(base + 9), sPre, Smat);
    k_smS<<<GRID(Bn * Nn)>>>(Smat);
    k_tax0<<<GRID(XSZ)>>>(Xin, Smat, tax0);
    k_zero_f<<<GRID(XSZ)>>>(tax1, XSZ);
    k_scatter<<<GRID(Bn * En * Hn * Tn)>>>(ei, deginv, Smat, tax0, tax1);
    k_sg<<<GRID(XSZ)>>>(tax0, tax1, F(base + 10), F(base + 11), sg);
    k_tconv<<<GRID(XSZ)>>>(sg, F(base + 12), F(base + 13), tcb);
    k_res<<<GRID(XSZ)>>>(Xin, F(base + 14), F(base + 15), resb);
    k_lnout<<<GRID(Bn * Tn * Nn)>>>(resb, tcb, F(base + 16), F(base + 17), Xout);
  };
  run_block(14, XA, XB);
  run_block(32, XB, XA);

  // final conv + node MLP + degree embedding -> nf (bf16)
  k_gf<<<GRID(Bn * Nn)>>>(XA, F(50), F(51), gf);
  k_nlin1<<<GRID(Bn * Nn * Hn)>>>(gf, F(52), F(53), nmA);
  k_ln_rows<<<GRID(Bn * Nn)>>>(nmA, F(54), F(55), Bn * Nn, Hn, 1);
  k_lin<<<GRID(Bn * Nn * Hn)>>>(nmA, F(56), F(57), nmB, Hn, Hn, Bn * Nn);
  k_ln_rows<<<GRID(Bn * Nn)>>>(nmB, F(58), F(59), Bn * Nn, Hn, 1);
  k_lin<<<GRID(Bn * Nn * Hn)>>>(nmB, F(60), F(61), nmA, Hn, Hn, Bn * Nn);
  k_ln_rows<<<GRID(Bn * Nn)>>>(nmA, F(62), F(63), Bn * Nn, Hn, 0);
  k_nf_build<<<GRID(Bn * Nn * Hn)>>>(nmA, F(74), deg, nfb);

  // bf16 copies of edge-MLP weights
  k_cvt_bf<<<GRID(128 * 128)>>>(F(64), w1b, 128 * 128);
  k_cvt_bf<<<GRID(64 * 128)>>>(F(68), w2b, 64 * 128);

  // fused all-pairs edge MLP (WMMA)
  k_edge<<<dim3(Nn / 16, Nn, Bn), dim3(32), 0, stream>>>(nfb, w1b, F(65), F(66), F(67), w2b, F(69),
                                                         F(70), F(71), F(72), F(73), dmean, Araw);

  // diagonal + symmetrize
  k_final<<<GRID(Bn * Nn * Nn)>>>(Araw, (float*)d_out);
#undef F
}
